// Model_87016037417183
// MI455X (gfx1250) — matrix operand order repaired and checked by reading, not compiled, not run on hardware
//
#include <hip/hip_runtime.h>
#include <hip/hip_bf16.h>

typedef __attribute__((ext_vector_type(2)))  float  v2f;
typedef __attribute__((ext_vector_type(8)))  float  v8f;
typedef __attribute__((ext_vector_type(16))) __bf16 v16bf;

#define WMMA4(a, b, c) \
  __builtin_amdgcn_wmma_f32_16x16x4_f32(false, (a), false, (b), (short)0, (c), false, false)
#define WMMA_BF16(a, b, c) \
  __builtin_amdgcn_wmma_f32_16x16x32_bf16(false, (a), false, (b), (short)0, (c), false, false)

__device__ __forceinline__ float sigf(float x) { return 1.0f / (1.0f + expf(-x)); }

// ---------------------------------------------------------------------------
// Utility: zero a float buffer
// ---------------------------------------------------------------------------
__global__ void zero_kernel(float* p, long n) {
  long i = (long)blockIdx.x * blockDim.x + threadIdx.x;
  if (i < n) p[i] = 0.0f;
}

// ---------------------------------------------------------------------------
// f32 -> bf16 conversion (RNE via compiler cvt)
// ---------------------------------------------------------------------------
__global__ void f32_to_bf16(const float* __restrict__ src, __bf16* __restrict__ dst, long n) {
  long i = (long)blockIdx.x * blockDim.x + threadIdx.x;
  if (i < n) dst[i] = (__bf16)src[i];
}

// ---------------------------------------------------------------------------
// Embedding gathers
// ---------------------------------------------------------------------------
__global__ void embed_in_kernel(const int* __restrict__ inputs,
                                const float* __restrict__ W, float* __restrict__ out) {
  long idx = (long)blockIdx.x * blockDim.x + threadIdx.x;   // 16*400*128
  if (idx >= 16L * 400 * 128) return;
  int e = (int)(idx & 127);
  long pos = idx >> 7;
  out[idx] = W[(long)inputs[pos] * 128 + e];
}

__global__ void embed_tgt_kernel(const int* __restrict__ target,
                                 const float* __restrict__ W, float* __restrict__ out) {
  long idx = (long)blockIdx.x * blockDim.x + threadIdx.x;   // 16*99*128
  if (idx >= 16L * 99 * 128) return;
  int e = (int)(idx & 127);
  long pos = idx >> 7;            // pos = b*99 + t
  int b = (int)(pos / 99), t = (int)(pos % 99);
  out[idx] = W[(long)target[b * 100 + t] * 128 + e];
}

// ---------------------------------------------------------------------------
// Generic f32 WMMA GEMM:  C(M,N) = A(M,K) @ W(N,K)^T + bias  (dims % 16 == 0)
// ---------------------------------------------------------------------------
__global__ void gemm_mk(const float* __restrict__ A, const float* __restrict__ W,
                        const float* __restrict__ bias, float* __restrict__ C,
                        int M, int N, int K) {
  const int wave = threadIdx.x >> 5;
  const int lane = threadIdx.x & 31;
  const int r = lane & 15;
  const int koff = (lane < 16) ? 0 : 2;
  const int mbase = (lane < 16) ? 0 : 8;
  const int mT = M >> 4, nT = N >> 4;
  const long total = (long)mT * nT;
  for (long t = (long)blockIdx.x * 8 + wave; t < total; t += (long)gridDim.x * 8) {
    const int mt = (int)(t / nT), nt = (int)(t % nT);
    const float* Ar = A + (long)(mt * 16 + r) * K;
    const float* Wr = W + (long)(nt * 16 + r) * K;
    v8f acc = {0, 0, 0, 0, 0, 0, 0, 0};
#pragma unroll 4
    for (int k = 0; k < K; k += 4) {
      v2f a, b;
      a.x = Ar[k + koff]; a.y = Ar[k + koff + 1];
      b.x = Wr[k + koff]; b.y = Wr[k + koff + 1];
      acc = WMMA4(a, b, acc);
    }
    const int col = nt * 16 + r;
    const float bv = bias ? bias[col] : 0.0f;
#pragma unroll
    for (int vi = 0; vi < 8; ++vi)
      C[(long)(mt * 16 + mbase + vi) * N + col] = acc[vi] + bv;
  }
}

// ---------------------------------------------------------------------------
// M=16 f32 WMMA GEMM (small N/K): C(16,N) = A(16,K) @ W(N,K)^T + bias
// ---------------------------------------------------------------------------
__global__ void gemm_m16(const float* __restrict__ A, const float* __restrict__ W,
                         const float* __restrict__ bias, float* __restrict__ C,
                         int N, int K) {
  __shared__ float As[16 * 768];
  for (int i = threadIdx.x; i < 16 * K; i += blockDim.x) As[i] = A[i];
  __syncthreads();
  const int wave = threadIdx.x >> 5, lane = threadIdx.x & 31;
  const int r = lane & 15;
  const int koff = (lane < 16) ? 0 : 2;
  const int mbase = (lane < 16) ? 0 : 8;
  const int nT = N >> 4;
  const float* Am = As + r * K;
  for (int t = blockIdx.x * 8 + wave; t < nT; t += gridDim.x * 8) {
    const float* Wr = W + (long)(t * 16 + r) * K;
    v8f acc = {0, 0, 0, 0, 0, 0, 0, 0};
#pragma unroll 4
    for (int k = 0; k < K; k += 4) {
      v2f a, b;
      a.x = Am[k + koff]; a.y = Am[k + koff + 1];
      b.x = Wr[k + koff]; b.y = Wr[k + koff + 1];
      acc = WMMA4(a, b, acc);
    }
    const int col = t * 16 + r;
    const float bv = bias ? bias[col] : 0.0f;
#pragma unroll
    for (int vi = 0; vi < 8; ++vi)
      C[(mbase + vi) * N + col] = acc[vi] + bv;
  }
}

// ---------------------------------------------------------------------------
// M=16 bf16 WMMA GEMM (dominant vocab projection):
// C(16,N) = A(16,K)bf16 @ W(N,K)bf16^T + bias, f32 accumulate.
// 16x16x32 bf16 WMMA: 2x less B-operand traffic and 8x fewer matrix ops vs
// f32 x4.  A (<=24KB bf16) cached in LDS.  One wave per 16x16 N-tile.
// ---------------------------------------------------------------------------
__global__ void gemm_m16_bf16(const __bf16* __restrict__ A, const __bf16* __restrict__ W,
                              const float* __restrict__ bias, float* __restrict__ C,
                              int N, int K) {
  __shared__ __bf16 As[16 * 768];
  for (int i = threadIdx.x; i < 16 * K; i += blockDim.x) As[i] = A[i];
  __syncthreads();
  const int wave = threadIdx.x >> 5, lane = threadIdx.x & 31;
  const int r = lane & 15;
  const int ka = (lane < 16) ? 0 : 8;    // unused: A is loaded with kb, like B (k = 16h + s)
  const int kb = (lane < 16) ? 0 : 16;   // B lane-half K offset
  const int mbase = (lane < 16) ? 0 : 8;
  const int nT = N >> 4;
  const __bf16* Am = As + r * K;
  for (int t = blockIdx.x * 8 + wave; t < nT; t += gridDim.x * 8) {
    const __bf16* Wr = W + (long)(t * 16 + r) * K;
    v8f acc = {0, 0, 0, 0, 0, 0, 0, 0};
    for (int k = 0; k < K; k += 32) {
      // prefetch the W stream ~512B ahead (global_prefetch_b8)
      __builtin_prefetch(Wr + k + kb + 256, 0, 0);
      v16bf a, b;
#pragma unroll
      for (int i = 0; i < 8; ++i) {
        a[i]     = Am[k + kb + i];        // K = k+kb .. k+kb+7
        a[i + 8] = Am[k + kb + 8 + i];    // K = k+kb+8 .. k+kb+15
      }
#pragma unroll
      for (int i = 0; i < 16; ++i) b[i] = Wr[k + kb + i];  // K = k+kb .. k+kb+15
      acc = WMMA_BF16(a, b, acc);
    }
    const int col = t * 16 + r;
    const float bv = bias ? bias[col] : 0.0f;
#pragma unroll
    for (int vi = 0; vi < 8; ++vi)
      C[(mbase + vi) * N + col] = acc[vi] + bv;
  }
}

// ---------------------------------------------------------------------------
// Persistent bidirectional encoder LSTM. grid=2 (dir), block=512 (16 waves).
// Wave w owns output columns [16w,16w+16) across all 4 gates -> c in registers.
// ---------------------------------------------------------------------------
__global__ void enc_lstm(const float* __restrict__ xWf, const float* __restrict__ xWb,
                         const float* __restrict__ Whhf, const float* __restrict__ Whhb,
                         const float* __restrict__ bihf, const float* __restrict__ bhhf,
                         const float* __restrict__ bihb, const float* __restrict__ bhhb,
                         float* __restrict__ encoded) {
  const int dir = blockIdx.x;
  const float* xW  = dir ? xWb  : xWf;
  const float* Whh = dir ? Whhb : Whhf;
  const float* bih = dir ? bihb : bihf;
  const float* bhh = dir ? bhhb : bhhf;
  __shared__ float h_lds[16 * 256];
  const int wave = threadIdx.x >> 5, lane = threadIdx.x & 31;
  const int col = (wave << 4) + (lane & 15);
  const int m = lane & 15;
  const int koff = (lane < 16) ? 0 : 2;
  const int mbase = (lane < 16) ? 0 : 8;
  for (int i = threadIdx.x; i < 16 * 256; i += blockDim.x) h_lds[i] = 0.0f;
  float creg[8];
#pragma unroll
  for (int vi = 0; vi < 8; ++vi) creg[vi] = 0.0f;
  float bsum[4];
#pragma unroll
  for (int g = 0; g < 4; ++g) bsum[g] = bih[g * 256 + col] + bhh[g * 256 + col];
  __syncthreads();

  for (int s = 0; s < 400; ++s) {
    const int t = dir ? (399 - s) : s;
    v8f acc[4];
#pragma unroll
    for (int g = 0; g < 4; ++g) acc[g] = (v8f){0, 0, 0, 0, 0, 0, 0, 0};
#pragma unroll 2
    for (int k = 0; k < 256; k += 4) {
      v2f a;
      a.x = h_lds[m * 256 + k + koff];
      a.y = h_lds[m * 256 + k + koff + 1];
#pragma unroll
      for (int g = 0; g < 4; ++g) {
        const float* wr = Whh + (long)(g * 256 + col) * 256;
        v2f b; b.x = wr[k + koff]; b.y = wr[k + koff + 1];
        acc[g] = WMMA4(a, b, acc[g]);
      }
    }
    __syncthreads();   // all waves finished reading h
#pragma unroll
    for (int vi = 0; vi < 8; ++vi) {
      const int row = mbase + vi;                       // batch index
      const long xbase = ((long)row * 400 + t) * 1024;
      float iv = acc[0][vi] + xW[xbase + 0   + col] + bsum[0];
      float fv = acc[1][vi] + xW[xbase + 256 + col] + bsum[1];
      float gv = acc[2][vi] + xW[xbase + 512 + col] + bsum[2];
      float ov = acc[3][vi] + xW[xbase + 768 + col] + bsum[3];
      float cv = sigf(fv) * creg[vi] + sigf(iv) * tanhf(gv);
      creg[vi] = cv;
      float hv = sigf(ov) * tanhf(cv);
      h_lds[row * 256 + col] = hv;
      encoded[((long)row * 400 + t) * 512 + dir * 256 + col] = hv;
    }
    __syncthreads();   // h fully updated before next step
  }
}

// ---------------------------------------------------------------------------
// Decoder LSTM cell (one step). 1 block, 512 threads (16 waves).
// ---------------------------------------------------------------------------
__global__ void dec_lstm(const float* __restrict__ embt,
                         const float* __restrict__ Wih, const float* __restrict__ Whh,
                         const float* __restrict__ bih, const float* __restrict__ bhh,
                         float* __restrict__ h, float* __restrict__ c) {
  const int wave = threadIdx.x >> 5, lane = threadIdx.x & 31;
  const int col = (wave << 4) + (lane & 15);
  const int m = lane & 15;
  const int koff = (lane < 16) ? 0 : 2;
  const int mbase = (lane < 16) ? 0 : 8;
  const long estr = 99 * 128;
  v8f acc[4];
#pragma unroll
  for (int g = 0; g < 4; ++g) acc[g] = (v8f){0, 0, 0, 0, 0, 0, 0, 0};
#pragma unroll 2
  for (int k = 0; k < 128; k += 4) {
    v2f a;
    a.x = embt[(long)m * estr + k + koff];
    a.y = embt[(long)m * estr + k + koff + 1];
#pragma unroll
    for (int g = 0; g < 4; ++g) {
      const float* wr = Wih + (long)(g * 256 + col) * 128;
      v2f b; b.x = wr[k + koff]; b.y = wr[k + koff + 1];
      acc[g] = WMMA4(a, b, acc[g]);
    }
  }
#pragma unroll 2
  for (int k = 0; k < 256; k += 4) {
    v2f a;
    a.x = h[m * 256 + k + koff];
    a.y = h[m * 256 + k + koff + 1];
#pragma unroll
    for (int g = 0; g < 4; ++g) {
      const float* wr = Whh + (long)(g * 256 + col) * 256;
      v2f b; b.x = wr[k + koff]; b.y = wr[k + koff + 1];
      acc[g] = WMMA4(a, b, acc[g]);
    }
  }
  __syncthreads();   // everyone done reading old h
  float bsum[4];
#pragma unroll
  for (int g = 0; g < 4; ++g) bsum[g] = bih[g * 256 + col] + bhh[g * 256 + col];
#pragma unroll
  for (int vi = 0; vi < 8; ++vi) {
    const int row = mbase + vi;
    float iv = acc[0][vi] + bsum[0];
    float fv = acc[1][vi] + bsum[1];
    float gv = acc[2][vi] + bsum[2];
    float ov = acc[3][vi] + bsum[3];
    float cv = sigf(fv) * c[row * 256 + col] + sigf(iv) * tanhf(gv);
    c[row * 256 + col] = cv;
    h[row * 256 + col] = sigf(ov) * tanhf(cv);
  }
}

// ---------------------------------------------------------------------------
// Attention + coverage + context + p_gen. grid=16 (one block per batch row).
// ---------------------------------------------------------------------------
__global__ void attn_step(const float* __restrict__ whenc,    // (16,400,256)
                          const float* __restrict__ wss,      // (16,256)
                          const float* __restrict__ encoded,  // (16,400,512)
                          const float* __restrict__ h,        // (16,256)
                          const float* __restrict__ embt,     // base + t*128, stride 99*128
                          const float* __restrict__ vw, const float* __restrict__ vb,
                          const float* __restrict__ whw, const float* __restrict__ whb,
                          const float* __restrict__ wslw, const float* __restrict__ wslb,
                          const float* __restrict__ wxw, const float* __restrict__ wxb,
                          float* __restrict__ cov, float* __restrict__ attn_out,
                          float* __restrict__ cat, float* __restrict__ pg,
                          float* __restrict__ clacc) {
  const int b = blockIdx.x;
  const int tid = threadIdx.x;            // 256 threads = 8 waves
  const int wave = tid >> 5, lane = tid & 31;
  __shared__ float logit[400];
  __shared__ float red[256];

  for (int l = wave; l < 400; l += 8) {
    const int rep = (b * 400 + l) & 15;
    const float* we = whenc + ((long)b * 400 + l) * 256;
    const float* ws = wss + rep * 256;
    float s = 0.0f;
    for (int k = lane; k < 256; k += 32) s += (we[k] + ws[k]) * vw[k];
#pragma unroll
    for (int off = 16; off > 0; off >>= 1) s += __shfl_xor(s, off, 32);
    if (lane == 0) logit[l] = s + vb[0];
  }
  __syncthreads();

  float mx = -1e30f;
  for (int l = tid; l < 400; l += 256) mx = fmaxf(mx, logit[l]);
  red[tid] = mx; __syncthreads();
  for (int st = 128; st > 0; st >>= 1) {
    if (tid < st) red[tid] = fmaxf(red[tid], red[tid + st]);
    __syncthreads();
  }
  mx = red[0]; __syncthreads();
  float sm = 0.0f;
  for (int l = tid; l < 400; l += 256) { float e = expf(logit[l] - mx); logit[l] = e; sm += e; }
  red[tid] = sm; __syncthreads();
  for (int st = 128; st > 0; st >>= 1) {
    if (tid < st) red[tid] += red[tid + st];
    __syncthreads();
  }
  const float inv = 1.0f / red[0]; __syncthreads();

  float cl = 0.0f;
  for (int l = tid; l < 400; l += 256) {
    float a = logit[l] * inv;
    logit[l] = a;
    attn_out[b * 400 + l] = a;
    float cv = cov[b * 400 + l];
    cl += fminf(a, cv);
    cov[b * 400 + l] = cv + a;
  }
  red[tid] = cl; __syncthreads();
  for (int st = 128; st > 0; st >>= 1) {
    if (tid < st) red[tid] += red[tid + st];
    __syncthreads();
  }
  if (tid == 0) clacc[b] += red[0];
  __syncthreads();

  float pgs = 0.0f;
  for (int j = tid; j < 512; j += 256) {
    const float* ep = encoded + (long)b * 400 * 512 + j;
    float s = 0.0f;
    for (int l = 0; l < 400; ++l) s += logit[l] * ep[(long)l * 512];
    cat[b * 768 + 256 + j] = s;
    pgs += s * whw[j];
  }
  pgs += h[b * 256 + tid] * wslw[tid];
  if (tid < 128) pgs += embt[(long)b * (99 * 128) + tid] * wxw[tid];
  red[tid] = pgs; __syncthreads();
  for (int st = 128; st > 0; st >>= 1) {
    if (tid < st) red[tid] += red[tid + st];
    __syncthreads();
  }
  if (tid == 0) pg[b] = sigf(red[0] + whb[0] + wslb[0] + wxb[0]);
  cat[b * 768 + tid] = h[b * 256 + tid];
}

// ---------------------------------------------------------------------------
// Per-row softmax stats over V=50000 (max, 1/sum). grid=16.
// ---------------------------------------------------------------------------
__global__ void row_stats(const float* __restrict__ x, float* __restrict__ rmax,
                          float* __restrict__ rinv) {
  const int b = blockIdx.x, tid = threadIdx.x;
  __shared__ float red[256];
  const float* row = x + (long)b * 50000;
  float mx = -1e30f;
  for (int i = tid; i < 50000; i += 256) mx = fmaxf(mx, row[i]);
  red[tid] = mx; __syncthreads();
  for (int st = 128; st > 0; st >>= 1) {
    if (tid < st) red[tid] = fmaxf(red[tid], red[tid + st]);
    __syncthreads();
  }
  mx = red[0]; __syncthreads();
  float s = 0.0f;
  for (int i = tid; i < 50000; i += 256) s += expf(row[i] - mx);
  red[tid] = s; __syncthreads();
  for (int st = 128; st > 0; st >>= 1) {
    if (tid < st) red[tid] += red[tid + st];
    __syncthreads();
  }
  if (tid == 0) { rmax[b] = mx; rinv[b] = 1.0f / red[0]; }
}

// ---------------------------------------------------------------------------
// p_w = p_gen * softmax(logits2) + 1/V, written to out[b, t, :].
// ---------------------------------------------------------------------------
__global__ void finalize(const float* __restrict__ lg2, const float* __restrict__ rmax,
                         const float* __restrict__ rinv, const float* __restrict__ pg,
                         float* __restrict__ out, int t) {
  long idx = (long)blockIdx.x * blockDim.x + threadIdx.x;   // 16*50000
  if (idx >= 16L * 50000) return;
  int b = (int)(idx / 50000), v = (int)(idx % 50000);
  float pv = expf(lg2[idx] - rmax[b]) * rinv[b];
  out[((long)b * 99 + t) * 50000 + v] = pg[b] * pv + (1.0f / 50000.0f);
}

// ---------------------------------------------------------------------------
// Pointer-copy scatter: out[b,t,inputs[b,l]] += (1-p_gen[b]) * attn[b,l]
// ---------------------------------------------------------------------------
__global__ void scatter_copy(const int* __restrict__ inputs, const float* __restrict__ attn,
                             const float* __restrict__ pg, float* __restrict__ out, int t) {
  int idx = blockIdx.x * blockDim.x + threadIdx.x;          // 16*400
  if (idx >= 16 * 400) return;
  int b = idx / 400;
  float w = (1.0f - pg[b]) * attn[idx];
  atomicAdd(&out[((long)b * 99 + t) * 50000 + inputs[idx]], w);
}

__global__ void write_covloss(const float* __restrict__ clacc, float* __restrict__ out) {
  if (threadIdx.x == 0 && blockIdx.x == 0) {
    float s = 0.0f;
    for (int i = 0; i < 16; ++i) s += clacc[i];
    out[0] = s;
  }
}

// ---------------------------------------------------------------------------
// Host orchestration
// ---------------------------------------------------------------------------
extern "C" void kernel_launch(void* const* d_in, const int* in_sizes, int n_in,
                              void* d_out, int out_size, void* d_ws, size_t ws_size,
                              hipStream_t stream) {
  (void)in_sizes; (void)n_in; (void)out_size; (void)ws_size;
  const int*   inputs  = (const int*)d_in[0];
  const int*   target  = (const int*)d_in[1];
  const float* embed_W = (const float*)d_in[2];
  const float* ef_Wih  = (const float*)d_in[3];
  const float* ef_Whh  = (const float*)d_in[4];
  const float* ef_bih  = (const float*)d_in[5];
  const float* ef_bhh  = (const float*)d_in[6];
  const float* eb_Wih  = (const float*)d_in[7];
  const float* eb_Whh  = (const float*)d_in[8];
  const float* eb_bih  = (const float*)d_in[9];
  const float* eb_bhh  = (const float*)d_in[10];
  const float* dWih    = (const float*)d_in[11];
  const float* dWhh    = (const float*)d_in[12];
  const float* dbih    = (const float*)d_in[13];
  const float* dbhh    = (const float*)d_in[14];
  const float* Wh_w    = (const float*)d_in[15];
  const float* Wh_b    = (const float*)d_in[16];
  const float* Ws_w    = (const float*)d_in[17];
  const float* Ws_b    = (const float*)d_in[18];
  const float* v_w     = (const float*)d_in[19];
  const float* v_b     = (const float*)d_in[20];
  const float* wh_w    = (const float*)d_in[21];
  const float* wh_b    = (const float*)d_in[22];
  const float* wsl_w   = (const float*)d_in[23];
  const float* wsl_b   = (const float*)d_in[24];
  const float* wx_w    = (const float*)d_in[25];
  const float* wx_b    = (const float*)d_in[26];
  const float* V1_w    = (const float*)d_in[27];
  const float* V1_b    = (const float*)d_in[28];
  const float* V2_w    = (const float*)d_in[29];
  const float* V2_b    = (const float*)d_in[30];
  float* out = (float*)d_out;

  // ---- workspace layout (floats). Buffers live through decoder first; the
  // encoder-only buffers (emb_in, xWf, xWb) sit at the end so the bf16 copy
  // of V2_w (38.4M bf16 = 19.2M float-slots) can alias them once they die.
  float* w = (float*)d_ws;
  float* encoded = w; w += 16L * 400 * 512;
  float* whenc   = w; w += 16L * 400 * 256;
  float* emb_tg  = w; w += 16L * 99 * 128;
  float* h_dec   = w; w += 16 * 256;
  float* c_dec   = w; w += 16 * 256;
  float* wss     = w; w += 16 * 256;
  float* attn    = w; w += 16 * 400;
  float* cov     = w; w += 16 * 400;
  float* cat     = w; w += 16 * 768;
  float* tmp1    = w; w += 16 * 768;
  float* lg2     = w; w += 16L * 50000;
  float* rmax    = w; w += 16;
  float* rinv    = w; w += 16;
  float* pg      = w; w += 16;
  float* clacc   = w; w += 16;
  float* tmp1bfF = w; w += (16 * 768) / 2 + 16;   // 12288 bf16
  __bf16* tmp1bf = (__bf16*)tmp1bfF;
  float* emb_in  = w; w += 16L * 400 * 128;
  float* xWf     = w; w += 16L * 400 * 1024;
  float* xWb     = w; w += 16L * 400 * 1024;
  // V2bf aliases [xWf ...): needs 19.2M float-slots, xWf+xWb provide 13.1M;
  // reserve the remainder explicitly.
  __bf16* V2bf   = (__bf16*)xWf;
  w += (50000L * 768 / 2) - 2L * 16 * 400 * 1024 + 64;

  // init state
  zero_kernel<<<(4096 + 255) / 256, 256, 0, stream>>>(h_dec, 4096);
  zero_kernel<<<(4096 + 255) / 256, 256, 0, stream>>>(c_dec, 4096);
  zero_kernel<<<(6400 + 255) / 256, 256, 0, stream>>>(cov, 6400);
  zero_kernel<<<1, 32, 0, stream>>>(clacc, 16);

  // embeddings
  embed_in_kernel<<<(16 * 400 * 128 + 255) / 256, 256, 0, stream>>>(inputs, embed_W, emb_in);
  embed_tgt_kernel<<<(16 * 99 * 128 + 255) / 256, 256, 0, stream>>>(target, embed_W, emb_tg);

  // hoisted input projections: (6400 x 1024, K=128) per direction
  gemm_mk<<<3200, 256, 0, stream>>>(emb_in, ef_Wih, nullptr, xWf, 6400, 1024, 128);
  gemm_mk<<<3200, 256, 0, stream>>>(emb_in, eb_Wih, nullptr, xWb, 6400, 1024, 128);

  // persistent bidirectional encoder (block 0 = fwd, block 1 = bwd)
  enc_lstm<<<2, 512, 0, stream>>>(xWf, xWb, ef_Whh, eb_Whh,
                                  ef_bih, ef_bhh, eb_bih, eb_bhh, encoded);

  // wh_enc = encoded @ Wh_w^T + Wh_b : (6400 x 256, K=512)
  gemm_mk<<<800, 256, 0, stream>>>(encoded, Wh_w, Wh_b, whenc, 6400, 256, 512);

  // xWf/xWb now dead -> materialize bf16 copy of V2_w over that region
  // (76.8 MB, L2-resident for the whole decoder scan)
  f32_to_bf16<<<(int)((50000L * 768 + 255) / 256), 256, 0, stream>>>(V2_w, V2bf, 50000L * 768);

  // decoder scan
  for (int t = 0; t < 99; ++t) {
    dec_lstm<<<1, 512, 0, stream>>>(emb_tg + t * 128, dWih, dWhh, dbih, dbhh, h_dec, c_dec);
    gemm_m16<<<2, 256, 0, stream>>>(h_dec, Ws_w, Ws_b, wss, 256, 256);
    attn_step<<<16, 256, 0, stream>>>(whenc, wss, encoded, h_dec, emb_tg + t * 128,
                                      v_w, v_b, wh_w, wh_b, wsl_w, wsl_b, wx_w, wx_b,
                                      cov, attn, cat, pg, clacc);
    gemm_m16<<<6, 256, 0, stream>>>(cat, V1_w, V1_b, tmp1, 768, 768);
    f32_to_bf16<<<(12288 + 255) / 256, 256, 0, stream>>>(tmp1, tmp1bf, 12288);
    // dominant GEMM: 16 x 50000, K=768, bf16 operands / f32 accumulate
    gemm_m16_bf16<<<391, 256, 0, stream>>>(tmp1bf, V2bf, V2_b, lg2, 50000, 768);
    row_stats<<<16, 256, 0, stream>>>(lg2, rmax, rinv);
    finalize<<<(16 * 50000 + 255) / 256, 256, 0, stream>>>(lg2, rmax, rinv, pg, out, t);
    scatter_copy<<<(16 * 400 + 255) / 256, 256, 0, stream>>>(inputs, attn, pg, out, t);
  }
  write_covloss<<<1, 32, 0, stream>>>(clacc, out + 16L * 99 * 50000);
}
